// NerUnit_40209483825381
// MI455X (gfx1250) — matrix operand order repaired and checked by reading, not compiled, not run on hardware
//
#include <hip/hip_runtime.h>

// ---------------------------------------------------------------------------
// NER biaffine unit, fused for MI455X (gfx1250, wave32, WMMA bf16).
// H=768, L=192, B=4, NUM_LABELS=16.
// Round 2: software-pipelined (register ping-pong) fragment loads so WMMAs
// wait on partial loadcnt instead of draining to 0 every k-step.
// ---------------------------------------------------------------------------

#define Hdim 768
#define Ldim 192
#define Bdim 4
#define NLAB 16
#define Rdim (Ldim * Bdim)   // 768 rows for stage-1 GEMMs

typedef __bf16 bf16_t;
typedef bf16_t v16bf __attribute__((ext_vector_type(16)));
typedef bf16_t v8bf  __attribute__((ext_vector_type(8)));
typedef float  v8f   __attribute__((ext_vector_type(8)));

union BFrag {
  v16bf  v;
  v8bf   q[2];
  bf16_t h[16];
};

// ------------------------------------------------------------------ converts
__global__ void cvt_f32_bf16(const float* __restrict__ src,
                             bf16_t* __restrict__ dst, int n) {
  int i = blockIdx.x * 256 + threadIdx.x;
  if (i < n) dst[i] = (bf16_t)src[i];
}

// ------------------------------------------------------- stage 1: 4 fused GEMMs
// a = h_ner*W_st^T, c = h_ner*W_en^T, gpre = tanh(h_share*Wn_s^T + h_ner*Wn_r^T + bn)
// One wave per 16x16 output tile; 48x48 tiles over (768 x 768).

struct Frags { BFrag fn, fsh, fst, fen, fns, fnr; };

__device__ __forceinline__ void load_frags(
    Frags& F, const bf16_t* hnRow, const bf16_t* hsRow,
    const bf16_t* wstRow, const bf16_t* wenRow,
    const bf16_t* wnsRow, const bf16_t* wnrRow, int kb, int half)
{
  const int ka = kb + 16 * half;           // A frag: contiguous 16 along k
  F.fn.q[0]  = *(const v8bf*)(hnRow + ka);
  F.fn.q[1]  = *(const v8bf*)(hnRow + ka + 8);
  F.fsh.q[0] = *(const v8bf*)(hsRow + ka);
  F.fsh.q[1] = *(const v8bf*)(hsRow + ka + 8);
  const int kw = kb + 16 * half;           // B frag: contiguous 16 along k
  F.fst.q[0] = *(const v8bf*)(wstRow + kw);  F.fst.q[1] = *(const v8bf*)(wstRow + kw + 8);
  F.fen.q[0] = *(const v8bf*)(wenRow + kw);  F.fen.q[1] = *(const v8bf*)(wenRow + kw + 8);
  F.fns.q[0] = *(const v8bf*)(wnsRow + kw);  F.fns.q[1] = *(const v8bf*)(wnsRow + kw + 8);
  F.fnr.q[0] = *(const v8bf*)(wnrRow + kw);  F.fnr.q[1] = *(const v8bf*)(wnrRow + kw + 8);
}

__device__ __forceinline__ void wmma4(const Frags& F, v8f& accA, v8f& accC, v8f& accG)
{
  accA = __builtin_amdgcn_wmma_f32_16x16x32_bf16(false, F.fn.v,  false, F.fst.v, (short)0, accA, false, false);
  accC = __builtin_amdgcn_wmma_f32_16x16x32_bf16(false, F.fn.v,  false, F.fen.v, (short)0, accC, false, false);
  accG = __builtin_amdgcn_wmma_f32_16x16x32_bf16(false, F.fsh.v, false, F.fns.v, (short)0, accG, false, false);
  accG = __builtin_amdgcn_wmma_f32_16x16x32_bf16(false, F.fn.v,  false, F.fnr.v, (short)0, accG, false, false);
}

__global__ __launch_bounds__(128) void stage1_kernel(
    const bf16_t* __restrict__ hn, const bf16_t* __restrict__ hs,
    const bf16_t* __restrict__ Wn, const bf16_t* __restrict__ W1,
    const float* __restrict__ bn,
    float* __restrict__ Abuf, float* __restrict__ Cbuf, float* __restrict__ Gbuf)
{
  const int lane  = threadIdx.x & 31;
  const int tile  = blockIdx.x * 4 + (threadIdx.x >> 5);   // 2304 tiles
  const int mbase = (tile / 48) * 16;
  const int obase = (tile % 48) * 16;
  const int m16   = lane & 15;
  const int half  = lane >> 4;

  const bf16_t* hnRow = hn + (size_t)(mbase + m16) * Hdim;
  const bf16_t* hsRow = hs + (size_t)(mbase + m16) * Hdim;
  const int     ocol  = obase + m16;
  const bf16_t* wstRow = W1 + (size_t)ocol * (3 * Hdim);        // W1[o, 0:H]
  const bf16_t* wenRow = wstRow + Hdim;                         // W1[o, H:2H]
  const bf16_t* wnsRow = Wn + (size_t)ocol * (2 * Hdim);        // Wn[o, 0:H]
  const bf16_t* wnrRow = wnsRow + Hdim;                         // Wn[o, H:2H]

  v8f accA = {}; v8f accC = {}; v8f accG = {};

  Frags F0, F1;
  load_frags(F0, hnRow, hsRow, wstRow, wenRow, wnsRow, wnrRow, 0, half);
  #pragma unroll 1
  for (int kb = 0; kb < Hdim; kb += 64) {
    load_frags(F1, hnRow, hsRow, wstRow, wenRow, wnsRow, wnrRow, kb + 32, half);
    wmma4(F0, accA, accC, accG);            // waits only for F0's loads
    if (kb + 64 < Hdim)
      load_frags(F0, hnRow, hsRow, wstRow, wenRow, wnsRow, wnrRow, kb + 64, half);
    wmma4(F1, accA, accC, accG);            // waits only for F1's loads
  }

  const float bno = bn[ocol];
  #pragma unroll
  for (int v = 0; v < 8; ++v) {            // C/D layout: row = v + 8*half, col = m16
    const int row = mbase + v + 8 * half;
    Abuf[(size_t)row * Hdim + ocol] = accA[v];
    Cbuf[(size_t)row * Hdim + ocol] = accC[v];
    Gbuf[(size_t)row * Hdim + ocol] = tanhf(accG[v] + bno);
  }
}

// ------------------------------------------------------- stage 2: g = max over L
__global__ void gmax_kernel(const float* __restrict__ Gbuf, float* __restrict__ g) {
  int idx = blockIdx.x * 256 + threadIdx.x;   // 3072 = b*768 + o
  int b = idx / Hdim, o = idx % Hdim;
  float mx = -3.4028234663852886e38f;
  for (int l = 0; l < Ldim; ++l)
    mx = fmaxf(mx, Gbuf[(size_t)(l * Bdim + b) * Hdim + o]);
  g[idx] = mx;
}

// ------------------------------------------------------- stage 2b: d = g * W_g^T (f32, tiny)
__global__ void dproj_kernel(const float* __restrict__ g, const float* __restrict__ W1,
                             float* __restrict__ dbuf) {
  int idx = blockIdx.x * 256 + threadIdx.x;   // 3072 = b*768 + o
  int b = idx / Hdim, o = idx % Hdim;
  const float* gr = g + (size_t)b * Hdim;
  const float* wr = W1 + (size_t)o * (3 * Hdim) + 2 * Hdim;   // W1[o, 2H:3H]
  float s = 0.f;
  for (int h = 0; h < Hdim; h += 4) {
    float4 gv = *(const float4*)(gr + h);
    float4 wv = *(const float4*)(wr + h);
    s += gv.x * wv.x + gv.y * wv.y + gv.z * wv.z + gv.w * wv.w;
  }
  dbuf[idx] = s;
}

// ------------------------------------------------------- stage 3: fused pre+LN+ELU+proj
// One wave per 16-row tile (fixed i, 4 j's, 4 b's) x 16 labels.
// pre never touches memory: stats pass + value pass feed WMMA A-frags directly.

struct RowCtx {
  const float* Ar; const float* Cr; const float* Dr;
  const float* b1; const float* gamma; const float* beta;
  float mean, rstd;
};

__device__ __forceinline__ void compute_x_frag(BFrag& fx, const RowCtx& R, int ka)
{
  #pragma unroll
  for (int piece = 0; piece < 2; ++piece) {       // chunks [ka,+8) and [ka+16,+8)
    #pragma unroll
    for (int q4 = 0; q4 < 2; ++q4) {
      const int o = ka + piece * 16 + q4 * 4;
      float4 av = *(const float4*)(R.Ar + o);
      float4 cv = *(const float4*)(R.Cr + o);
      float4 dv = *(const float4*)(R.Dr + o);
      float4 bv = *(const float4*)(R.b1 + o);
      float4 gv = *(const float4*)(R.gamma + o);
      float4 ev = *(const float4*)(R.beta + o);
      float p[4]  = { av.x + cv.x + dv.x + bv.x, av.y + cv.y + dv.y + bv.y,
                      av.z + cv.z + dv.z + bv.z, av.w + cv.w + dv.w + bv.w };
      float gm[4] = { gv.x, gv.y, gv.z, gv.w };
      float be[4] = { ev.x, ev.y, ev.z, ev.w };
      #pragma unroll
      for (int e = 0; e < 4; ++e) {
        float xn = (p[e] - R.mean) * R.rstd * gm[e] + be[e];
        float el = xn > 0.f ? xn : (expf(xn) - 1.f);   // ELU(alpha=1)
        fx.h[piece * 8 + q4 * 4 + e] = (bf16_t)el;
      }
    }
  }
}

__global__ __launch_bounds__(128) void stage3_kernel(
    const float* __restrict__ Abuf, const float* __restrict__ Cbuf,
    const float* __restrict__ dbuf, const float* __restrict__ b1,
    const float* __restrict__ gamma, const float* __restrict__ beta,
    const bf16_t* __restrict__ WtB, const float* __restrict__ bt,
    const float* __restrict__ mask, float* __restrict__ out)
{
  const int lane  = threadIdx.x & 31;
  const int tile  = blockIdx.x * 4 + (threadIdx.x >> 5);   // 9216 = 192 * 48
  const int i     = tile / 48;
  const int jbase = (tile % 48) * 4;
  const int m16   = lane & 15;
  const int half  = lane >> 4;
  const int rbase = (i * Ldim + jbase) * Bdim;

  if (jbase + 3 < i) {                       // tile fully below diagonal -> zeros
    #pragma unroll
    for (int v = 0; v < 8; ++v)
      out[(size_t)(rbase + v + 8 * half) * NLAB + m16] = 0.f;
    return;
  }

  // This lane's A-frag row: m16 = jj*4 + bb
  const int jj = m16 >> 2, bb = m16 & 3;
  const int j  = jbase + jj;
  RowCtx R;
  R.Ar = Abuf + (size_t)(i * Bdim + bb) * Hdim;
  R.Cr = Cbuf + (size_t)(j * Bdim + bb) * Hdim;
  R.Dr = dbuf + (size_t)bb * Hdim;
  R.b1 = b1; R.gamma = gamma; R.beta = beta;

  // ---- pass 1: LN stats (each half-lane does 384 cols, combine via shfl_xor 16)
  float s1 = 0.f, s2 = 0.f;
  for (int o = half * 384; o < half * 384 + 384; o += 4) {
    float4 av = *(const float4*)(R.Ar + o);
    float4 cv = *(const float4*)(R.Cr + o);
    float4 dv = *(const float4*)(R.Dr + o);
    float4 bv = *(const float4*)(b1 + o);
    float p0 = av.x + cv.x + dv.x + bv.x, p1 = av.y + cv.y + dv.y + bv.y;
    float p2 = av.z + cv.z + dv.z + bv.z, p3 = av.w + cv.w + dv.w + bv.w;
    s1 += (p0 + p1) + (p2 + p3);
    s2 += (p0 * p0 + p1 * p1) + (p2 * p2 + p3 * p3);
  }
  s1 += __shfl_xor(s1, 16, 32);
  s2 += __shfl_xor(s2, 16, 32);
  R.mean = s1 * (1.f / 768.f);
  const float var = s2 * (1.f / 768.f) - R.mean * R.mean;
  R.rstd = rsqrtf(var + 1e-5f);

  // ---- pass 2: x -> bf16 A-frags, WMMA vs Wt; ping-pong the Wt B-fragment so
  // its loads overlap the (long) VALU chain producing the next x fragment.
  const bf16_t* WtRow = WtB + (size_t)m16 * Hdim;   // B frag: lane col = label m16
  const int kwoff = 8 * half;
  v8f acc = {};
  BFrag fw0, fw1;
  fw0.q[0] = *(const v8bf*)(WtRow + kwoff);
  fw0.q[1] = *(const v8bf*)(WtRow + kwoff + 16);
  #pragma unroll 1
  for (int kb = 0; kb < Hdim; kb += 64) {
    fw1.q[0] = *(const v8bf*)(WtRow + kb + 32 + kwoff);
    fw1.q[1] = *(const v8bf*)(WtRow + kb + 32 + kwoff + 16);
    BFrag fx;
    compute_x_frag(fx, R, kb + 8 * half);
    acc = __builtin_amdgcn_wmma_f32_16x16x32_bf16(false, fx.v, false, fw0.v, (short)0, acc, false, false);
    if (kb + 64 < Hdim) {
      fw0.q[0] = *(const v8bf*)(WtRow + kb + 64 + kwoff);
      fw0.q[1] = *(const v8bf*)(WtRow + kb + 64 + kwoff + 16);
    }
    compute_x_frag(fx, R, kb + 32 + 8 * half);
    acc = __builtin_amdgcn_wmma_f32_16x16x32_bf16(false, fx.v, false, fw1.v, (short)0, acc, false, false);
  }

  // ---- epilogue: +bt, sigmoid, mask * triu
  const float btn = bt[m16];
  #pragma unroll
  for (int v = 0; v < 8; ++v) {
    const int mloc = v + 8 * half;           // D row within tile
    const int bbv  = mloc & 3;
    const int jv   = jbase + (mloc >> 2);
    float z  = acc[v] + btn;
    float sg = 1.f / (1.f + expf(-z));
    float mk = (jv >= i) ? mask[i * Bdim + bbv] * mask[jv * Bdim + bbv] : 0.f;
    out[(size_t)(rbase + mloc) * NLAB + m16] = sg * mk;
  }
}

// ---------------------------------------------------------------------------
extern "C" void kernel_launch(void* const* d_in, const int* in_sizes, int n_in,
                              void* d_out, int out_size, void* d_ws, size_t ws_size,
                              hipStream_t stream)
{
  (void)in_sizes; (void)n_in; (void)out_size; (void)ws_size;
  const float* h_ner   = (const float*)d_in[0];
  const float* h_share = (const float*)d_in[1];
  const float* mask    = (const float*)d_in[2];
  const float* Wn      = (const float*)d_in[3];
  const float* bn      = (const float*)d_in[4];
  const float* W1      = (const float*)d_in[5];
  const float* b1      = (const float*)d_in[6];
  const float* gamma   = (const float*)d_in[7];
  const float* beta    = (const float*)d_in[8];
  const float* Wt      = (const float*)d_in[9];
  const float* bt      = (const float*)d_in[10];
  float* out = (float*)d_out;

  char* ws = (char*)d_ws;
  size_t off = 0;
  auto take = [&](size_t bytes) -> char* {
    char* p = ws + off;
    off = (off + bytes + 255) & ~(size_t)255;
    return p;
  };
  const size_t NRH = (size_t)Rdim * Hdim;          // 589824
  bf16_t* hn_bf = (bf16_t*)take(NRH * 2);
  bf16_t* hs_bf = (bf16_t*)take(NRH * 2);
  bf16_t* Wn_bf = (bf16_t*)take((size_t)Hdim * 2 * Hdim * 2);
  bf16_t* W1_bf = (bf16_t*)take((size_t)Hdim * 3 * Hdim * 2);
  bf16_t* Wt_bf = (bf16_t*)take((size_t)NLAB * Hdim * 2);
  float*  Abuf  = (float*)take(NRH * 4);
  float*  Cbuf  = (float*)take(NRH * 4);
  float*  Gbuf  = (float*)take(NRH * 4);
  float*  gbuf  = (float*)take((size_t)Bdim * Hdim * 4);
  float*  dbuf  = (float*)take((size_t)Bdim * Hdim * 4);

  const int nWn = Hdim * 2 * Hdim;     // 1179648
  const int nW1 = Hdim * 3 * Hdim;     // 1769472
  const int nWt = NLAB * Hdim;         // 12288

  cvt_f32_bf16<<<(int)((NRH + 255) / 256), 256, 0, stream>>>(h_ner,  hn_bf, (int)NRH);
  cvt_f32_bf16<<<(int)((NRH + 255) / 256), 256, 0, stream>>>(h_share, hs_bf, (int)NRH);
  cvt_f32_bf16<<<(nWn + 255) / 256, 256, 0, stream>>>(Wn, Wn_bf, nWn);
  cvt_f32_bf16<<<(nW1 + 255) / 256, 256, 0, stream>>>(W1, W1_bf, nW1);
  cvt_f32_bf16<<<(nWt + 255) / 256, 256, 0, stream>>>(Wt, Wt_bf, nWt);

  stage1_kernel<<<576, 128, 0, stream>>>(hn_bf, hs_bf, Wn_bf, W1_bf, bn, Abuf, Cbuf, Gbuf);
  gmax_kernel<<<12, 256, 0, stream>>>(Gbuf, gbuf);
  dproj_kernel<<<12, 256, 0, stream>>>(gbuf, W1, dbuf);
  stage3_kernel<<<2304, 128, 0, stream>>>(Abuf, Cbuf, dbuf, b1, gamma, beta,
                                          Wt_bf, bt, mask, out);
}
